// CustomModel_71227737637112
// MI455X (gfx1250) — matrix operand order repaired and checked by reading, not compiled, not run on hardware
//
#include <hip/hip_runtime.h>
#include <hip/hip_bf16.h>

// ---------------------------------------------------------------------------
// Problem: B=64 batches, N=128 points, D=64 dims.
//   dist[b,i,j] = || y_true[b,i] - y_pred[b,j] ||_2
//   col[b]      = linear_sum_assignment(dist[b])       (Jonker-Volgenant)
//   out         = mean_b( sum_i dist[b,i,col[b,i]] )
// ---------------------------------------------------------------------------

#define B_  64
#define N_  128
#define D_  64
#define PAD 72   // LDS row stride in halves (144 B: 16-B aligned, bank-spread)

typedef __attribute__((ext_vector_type(16))) _Float16 v16h;
typedef __attribute__((ext_vector_type(8)))  float    v8f;

union AFrag { v16h v; _Float16 e[16]; };
struct __align__(16) H8 { _Float16 e[8]; };

// ---------------------------------------------------------------------------
// Kernel 1: pairwise distances via WMMA.
//   One block per batch, 256 threads = 8 waves; wave w owns output rows
//   [16w,16w+16). Gram matrix G = T * P^T with v_wmma_f32_16x16x32_f16 and
//   an f32 -> (hi,lo) f16 split: t.p ~= t_hi.p_hi + t_hi.p_lo + t_lo.p_hi.
//   y_pred is split ONCE per block into LDS (hi/lo f16); B fragments are
//   then two ds_load_b128 per v16h (lane's 16 elems are k runs 8h..8h+7, 16+8h..16+8h+7 in the
//   ISA 16-bit B 32x16 layout).
// ---------------------------------------------------------------------------
__global__ __launch_bounds__(256) void dist_kernel(
    const float* __restrict__ yt, const float* __restrict__ yp,
    float* __restrict__ dist) {
  const int b = blockIdx.x;
  const int t = threadIdx.x;

  __shared__ _Float16 sp_hi[N_][PAD];
  __shared__ _Float16 sp_lo[N_][PAD];
  __shared__ float tnorm[N_], pnorm[N_];

  // ---- squared row norms ----
  {
    const int r = t & (N_ - 1);
    const float* src = ((t < N_) ? yt : yp) + (b * N_ + r) * D_;
    float s = 0.f;
#pragma unroll
    for (int d = 0; d < D_; ++d) { float x = src[d]; s += x * x; }
    if (t < N_) tnorm[r] = s; else pnorm[r] = s;
  }

  // ---- stage y_pred into LDS as hi/lo f16 (once per block) ----
  {
    const float4* src = (const float4*)(yp + (size_t)b * N_ * D_);
    for (int idx = t; idx < N_ * D_ / 4; idx += 256) {
      float4 x = src[idx];
      const int n = idx >> 4;            // row (D_/4 = 16 float4 per row)
      const int k = (idx & 15) * 4;      // k offset
      float c0 = x.x, c1 = x.y, c2 = x.z, c3 = x.w;
      _Float16 h0 = (_Float16)c0, h1 = (_Float16)c1,
               h2 = (_Float16)c2, h3 = (_Float16)c3;
      sp_hi[n][k + 0] = h0; sp_lo[n][k + 0] = (_Float16)(c0 - (float)h0);
      sp_hi[n][k + 1] = h1; sp_lo[n][k + 1] = (_Float16)(c1 - (float)h1);
      sp_hi[n][k + 2] = h2; sp_lo[n][k + 2] = (_Float16)(c2 - (float)h2);
      sp_hi[n][k + 3] = h3; sp_lo[n][k + 3] = (_Float16)(c3 - (float)h3);
    }
  }
  __syncthreads();

  const int wave = t >> 5;      // tile-row 0..7
  const int lane = t & 31;
  const int half = lane >> 4;   // 0 or 1
  const int l15  = lane & 15;

  // ---- A fragments (y_true rows), ISA 16-bit A 16x32 layout:
  //      lanes 0-15: M=lane, VGPR q<4 -> K=2q+s ; q>=4 -> K=16+2(q-4)+s
  //      lanes 16-31: same with K offset +8.  Wave-private -> split in regs.
  AFrag a_hi[2], a_lo[2];
  {
    const int am = wave * 16 + l15;
    const float* arow = yt + ((size_t)b * N_ + am) * D_;
#pragma unroll
    for (int ks = 0; ks < 2; ++ks) {
#pragma unroll
      for (int q = 0; q < 8; ++q) {
#pragma unroll
        for (int s = 0; s < 2; ++s) {
          int kl = (q < 4) ? (8 * half + 2 * q + s)
                           : (8 * half + 16 + 2 * (q - 4) + s);
          float x = arow[ks * 32 + kl];
          _Float16 hi = (_Float16)x;
          a_hi[ks].e[2 * q + s] = hi;
          a_lo[ks].e[2 * q + s] = (_Float16)(x - (float)hi);
        }
      }
    }
  }

  float* dmat = dist + (size_t)b * N_ * N_;

  for (int tn = 0; tn < 8; ++tn) {
    // ---- B fragments from LDS: elems 0-7 -> k=8h+s, elems 8-15 -> k=16+8h+(s-8) ----
    const int bn = tn * 16 + l15;
    AFrag b_hi[2], b_lo[2];
#pragma unroll
    for (int ks = 0; ks < 2; ++ks) {
      const int base = ks * 32 + 8 * half;
      *(H8*)&b_hi[ks].e[0] = *(const H8*)&sp_hi[bn][base];
      *(H8*)&b_hi[ks].e[8] = *(const H8*)&sp_hi[bn][base + 16];
      *(H8*)&b_lo[ks].e[0] = *(const H8*)&sp_lo[bn][base];
      *(H8*)&b_lo[ks].e[8] = *(const H8*)&sp_lo[bn][base + 16];
    }

    v8f c = {};
#pragma unroll
    for (int ks = 0; ks < 2; ++ks) {
      c = __builtin_amdgcn_wmma_f32_16x16x32_f16(false, a_hi[ks].v, false,
                                                 b_hi[ks].v, (short)0, c,
                                                 false, false);
      c = __builtin_amdgcn_wmma_f32_16x16x32_f16(false, a_hi[ks].v, false,
                                                 b_lo[ks].v, (short)0, c,
                                                 false, false);
      c = __builtin_amdgcn_wmma_f32_16x16x32_f16(false, a_lo[ks].v, false,
                                                 b_hi[ks].v, (short)0, c,
                                                 false, false);
    }

    // C/D layout: lanes 0-15 -> N=lane, M=r ; lanes 16-31 -> N=lane-16, M=r+8
#pragma unroll
    for (int r = 0; r < 8; ++r) {
      int m = wave * 16 + r + 8 * half;
      int n = tn * 16 + l15;
      float val = tnorm[m] + pnorm[n] - 2.0f * c[r];
      dmat[m * N_ + n] = sqrtf(fmaxf(val, 0.0f));
    }
  }
}

// ---------------------------------------------------------------------------
// Kernel 2: Jonker-Volgenant LSAP, one block (128 threads) per batch.
// e-maxx formulation matching the reference; inner scan parallel over
// columns; argmin is a first-index-tie-break tree reduction (== np.argmin).
// Internal math in f64 like the reference.
// ---------------------------------------------------------------------------
__global__ __launch_bounds__(N_) void lsap_kernel(
    const float* __restrict__ dist, int* __restrict__ col) {
  const int b = blockIdx.x;
  const int t = threadIdx.x;  // handles column j = t+1
  const float* cost = dist + (size_t)b * N_ * N_;

  __shared__ double u[N_ + 1], v[N_ + 1], minv[N_ + 1];
  __shared__ int    p[N_ + 1], way[N_ + 1];
  __shared__ unsigned char used[N_ + 1];
  __shared__ double red_v[N_];
  __shared__ int    red_j[N_];
  __shared__ int    sh_j0;

  u[t + 1] = 0.0; v[t + 1] = 0.0; p[t + 1] = 0; way[t + 1] = 0;
  if (t == 0) { u[0] = 0.0; v[0] = 0.0; p[0] = 0; way[0] = 0; }
  __syncthreads();

  for (int i = 1; i <= N_; ++i) {
    if (t == 0) { p[0] = i; sh_j0 = 0; used[0] = 0; minv[0] = 1e300; }
    minv[t + 1] = 1e300;
    used[t + 1] = 0;
    __syncthreads();

    while (true) {
      if (t == 0) used[sh_j0] = 1;
      __syncthreads();

      const int    j0  = sh_j0;
      const int    i0  = p[j0];
      const double ui0 = u[i0];
      const int    j   = t + 1;
      const bool   ju  = (used[j] != 0);

      if (!ju) {
        double cur = (double)cost[(i0 - 1) * N_ + (j - 1)] - ui0 - v[j];
        if (cur < minv[j]) { minv[j] = cur; way[j] = j0; }
      }
      red_v[t] = ju ? 1e300 : minv[j];
      red_j[t] = j;
      __syncthreads();

      for (int s = N_ / 2; s > 0; s >>= 1) {
        if (t < s) {
          double ov = red_v[t + s]; int oj = red_j[t + s];
          if (ov < red_v[t] || (ov == red_v[t] && oj < red_j[t])) {
            red_v[t] = ov; red_j[t] = oj;
          }
        }
        __syncthreads();
      }
      const double delta = red_v[0];
      const int    j1    = red_j[0];

      // u[p[used]] += delta (assigned rows of used cols distinct -> race-free)
      if (ju) { u[p[j]] += delta; v[j] -= delta; }
      else    { minv[j] -= delta; }
      if (t == 0) { u[p[0]] += delta; v[0] -= delta; sh_j0 = j1; }
      __syncthreads();

      if (p[j1] == 0) break;
    }

    if (t == 0) {  // augment alternating path
      int j0a = sh_j0;
      while (j0a != 0) { int j1a = way[j0a]; p[j0a] = p[j1a]; j0a = j1a; }
    }
    __syncthreads();
  }

  const int pj = p[t + 1];
  if (pj > 0) col[b * N_ + (pj - 1)] = t;  // col_of_row[p[j]-1] = j-1
}

// ---------------------------------------------------------------------------
// Kernel 3: per-batch matched-distance sum (deterministic tree reduction).
// ---------------------------------------------------------------------------
__global__ __launch_bounds__(N_) void gather_kernel(
    const float* __restrict__ dist, const int* __restrict__ col,
    float* __restrict__ bsum) {
  const int b = blockIdx.x, t = threadIdx.x;
  __shared__ float red[N_];
  const int c = col[b * N_ + t];
  red[t] = dist[(size_t)b * N_ * N_ + t * N_ + c];
  __syncthreads();
  for (int s = N_ / 2; s > 0; s >>= 1) {
    if (t < s) red[t] += red[t + s];
    __syncthreads();
  }
  if (t == 0) bsum[b] = red[0];
}

// Kernel 4: mean over batches -> scalar.
__global__ __launch_bounds__(B_) void final_kernel(
    const float* __restrict__ bsum, float* __restrict__ out) {
  const int t = threadIdx.x;
  __shared__ float red[B_];
  red[t] = bsum[t];
  __syncthreads();
  for (int s = B_ / 2; s > 0; s >>= 1) {
    if (t < s) red[t] += red[t + s];
    __syncthreads();
  }
  if (t == 0) out[0] = red[0] / (float)B_;
}

// ---------------------------------------------------------------------------
extern "C" void kernel_launch(void* const* d_in, const int* in_sizes, int n_in,
                              void* d_out, int out_size, void* d_ws,
                              size_t ws_size, hipStream_t stream) {
  (void)in_sizes; (void)n_in; (void)out_size; (void)ws_size;
  const float* y_true = (const float*)d_in[0];
  const float* y_pred = (const float*)d_in[1];

  // workspace layout
  float* dist = (float*)d_ws;                                   // 4 MiB
  int*   col  = (int*)((char*)d_ws + (size_t)B_ * N_ * N_ * 4); // 32 KiB
  float* bsum = (float*)((char*)col + (size_t)B_ * N_ * 4);     // 256 B

  dist_kernel  <<<B_, 256, 0, stream>>>(y_true, y_pred, dist);
  lsap_kernel  <<<B_, N_, 0, stream>>>(dist, col);
  gather_kernel<<<B_, N_, 0, stream>>>(dist, col, bsum);
  final_kernel <<<1, B_, 0, stream>>>(bsum, (float*)d_out);
}
